// MTP_29738353557787
// MI455X (gfx1250) — compile-verified, matrix operand order checked by reading, not run on hardware
//
#include <hip/hip_runtime.h>

typedef __attribute__((ext_vector_type(16))) __bf16 v16bf;
typedef __attribute__((ext_vector_type(8)))  __bf16 v8bf;
typedef __attribute__((ext_vector_type(8)))  float  v8f;

#define EPS_RMS 1e-8f
#define EPS_LN  1e-5f

// ---------------------------------------------------------------- embedding gather
__global__ void mtp_gather(const int* __restrict__ idx, const float* __restrict__ table,
                           float* __restrict__ emb, int D) {
  int r = blockIdx.x;
  long long src = (long long)idx[r] * D;
  long long dst = (long long)r * D;
  for (int j = threadIdx.x * 4; j < D; j += blockDim.x * 4) {
    *(float4*)(emb + dst + j) = *(const float4*)(table + src + j);
  }
}

// ---------------------------------------------------------------- block reduction (256 thr)
__device__ __forceinline__ float mtp_block_sum(float v, float* buf) {
  int t = threadIdx.x;
  buf[t] = v;
  __syncthreads();
  for (int s = 128; s > 0; s >>= 1) {
    if (t < s) buf[t] += buf[t + s];
    __syncthreads();
  }
  float r = buf[0];
  __syncthreads();
  return r;
}

// ---------------------------------------------------------------- RMS(h) || RMS(e) -> bf16
// row r -> b = r/I ; h row = r + b*hskip ; e row = r + b*eskip + eoff. D == 1024, 256 threads.
__global__ void mtp_rms_merge(const float* __restrict__ hsrc, int hskip,
                              const float* __restrict__ esrc, int eskip, int eoff,
                              __bf16* __restrict__ merged, int I, int D) {
  __shared__ float buf[256];
  int r = blockIdx.x;
  int b = r / I;
  const float* hrow = hsrc + (long long)(r + b * hskip) * D;
  const float* erow = esrc + (long long)(r + b * eskip + eoff) * D;
  __bf16* mrow = merged + (long long)r * (2 * D);
  int t = threadIdx.x;

  float v[4];
  float s = 0.f;
#pragma unroll
  for (int j = 0; j < 4; ++j) { float x = hrow[t + j * 256]; v[j] = x; s += x * x; }
  float scale = rsqrtf(mtp_block_sum(s, buf) / D + EPS_RMS);
#pragma unroll
  for (int j = 0; j < 4; ++j) mrow[t + j * 256] = (__bf16)(v[j] * scale);

  s = 0.f;
#pragma unroll
  for (int j = 0; j < 4; ++j) { float x = erow[t + j * 256]; v[j] = x; s += x * x; }
  scale = rsqrtf(mtp_block_sum(s, buf) / D + EPS_RMS);
#pragma unroll
  for (int j = 0; j < 4; ++j) mrow[D + t + j * 256] = (__bf16)(v[j] * scale);
}

// ---------------------------------------------------------------- y = LN(a + b)*g + bb  (fp32 + bf16)
__global__ void mtp_ln_res(const float* __restrict__ a, const float* __restrict__ bres,
                           const float* __restrict__ g, const float* __restrict__ bb,
                           float* __restrict__ out, __bf16* __restrict__ outbf, int D) {
  __shared__ float buf[256];
  int r = blockIdx.x;
  int t = threadIdx.x;
  const float* arow = a + (long long)r * D;
  const float* brow = bres + (long long)r * D;
  float v[4];
  float s = 0.f;
#pragma unroll
  for (int j = 0; j < 4; ++j) {
    float x = arow[t + j * 256] + brow[t + j * 256];
    v[j] = x; s += x;
  }
  float mean = mtp_block_sum(s, buf) / D;
  s = 0.f;
#pragma unroll
  for (int j = 0; j < 4; ++j) { float d = v[j] - mean; s += d * d; }
  float inv = rsqrtf(mtp_block_sum(s, buf) / D + EPS_LN);
  float* orow = out + (long long)r * D;
  __bf16* ob = outbf + (long long)r * D;
#pragma unroll
  for (int j = 0; j < 4; ++j) {
    int c = t + j * 256;
    float y = (v[j] - mean) * inv * g[c] + bb[c];
    orow[c] = y;
    ob[c] = (__bf16)y;
  }
}

// ---------------------------------------------------------------- WMMA bf16 GEMM
// C[M,N] = A_bf16[M,Kd] * Bw_fp32[N,Kd]^T (+bias) (+relu). fp32 out (row stride ldC)
// plus optional bf16 mirror (row stride N).
// 256 thr = 8 waves; 128x128 block tile; each wave: 32x64 = 2x4
// v_wmma_f32_16x16x32_bf16 accumulators (8 WMMA per K-step, A-fragments reused 4x).
__global__ void __launch_bounds__(256)
mtp_gemm(const __bf16* __restrict__ A, const float* __restrict__ Bw,
         const float* __restrict__ bias, float* __restrict__ C,
         __bf16* __restrict__ Cbf, int M, int N, int Kd, long long ldC, int relu) {
  __shared__ __attribute__((aligned(16))) __bf16 As[128][40];  // 80B rows: 16B-aligned, bank-skewed
  __shared__ __attribute__((aligned(16))) __bf16 Bs[128][40];

  int tid  = threadIdx.x;
  int wave = tid >> 5, lane = tid & 31;
  int wr = wave >> 1, wc = wave & 1;        // 4x2 wave grid: rows of 32, cols of 64
  int lm = lane & 15, kh = lane >> 4;       // WMMA lane decomposition

  int bm = blockIdx.y * 128, bn = blockIdx.x * 128;

  v8f acc[2][4];
#pragma unroll
  for (int i = 0; i < 2; ++i)
#pragma unroll
    for (int j = 0; j < 4; ++j)
#pragma unroll
      for (int e = 0; e < 8; ++e) acc[i][j][e] = 0.f;

  int lrow  = tid >> 1;                     // 2 threads per 128-long tile row
  int lhalf = (tid & 1) * 16;               // each covers 16 of the 32 K slots
  int gm = bm + lrow;
  int gn = bn + lrow;

  const __bf16* aRow = A  + (long long)gm * Kd + lhalf;
  const float*  bRow = Bw + (long long)gn * Kd + lhalf;

  for (int k0 = 0; k0 < Kd; k0 += 32) {
    // stage A (bf16, 2x 16B loads per thread)
    {
      v8bf a0, a1;
      if (gm < M) {
        const __bf16* ap = aRow + k0;
        a0 = *(const v8bf*)ap;
        a1 = *(const v8bf*)(ap + 8);
      } else {
#pragma unroll
        for (int e = 0; e < 8; ++e) { a0[e] = (__bf16)0.f; a1[e] = (__bf16)0.f; }
      }
      *(v8bf*)&As[lrow][lhalf]     = a0;
      *(v8bf*)&As[lrow][lhalf + 8] = a1;
    }
    // stage B (fp32 weights -> bf16 in LDS) + prefetch next K-step's weight segment
    {
      const float* bp = bRow + k0;
#pragma unroll
      for (int q = 0; q < 16; q += 4) {
        float4 f;
        if (gn < N) f = *(const float4*)(bp + q);
        else        f = make_float4(0.f, 0.f, 0.f, 0.f);
        __bf16* d = &Bs[lrow][lhalf + q];
        d[0] = (__bf16)f.x; d[1] = (__bf16)f.y; d[2] = (__bf16)f.z; d[3] = (__bf16)f.w;
      }
      if (gn < N && k0 + 32 < Kd) __builtin_prefetch(bp + 32, 0, 1);  // global_prefetch_b8
    }
    __syncthreads();

    // fragments: ISA 16-bit A layout = two contiguous 8-elem (16B) runs per lane:
    //   VGPR0-3: K = 8*kh .. 8*kh+7 ; VGPR4-7: K = 16+8*kh .. 16+8*kh+7
    v16bf afrag[2], bfrag[4];
#pragma unroll
    for (int tI = 0; tI < 2; ++tI) {
      const __bf16* pa = &As[wr * 32 + tI * 16 + lm][kh * 8];
      v8bf lo = *(const v8bf*)pa;
      v8bf hi = *(const v8bf*)(pa + 16);
#pragma unroll
      for (int e = 0; e < 8; ++e) { afrag[tI][e] = lo[e]; afrag[tI][e + 8] = hi[e]; }
    }
#pragma unroll
    for (int tI = 0; tI < 4; ++tI) {
      const __bf16* pb = &Bs[wc * 64 + tI * 16 + lm][kh * 8];
      v8bf lo = *(const v8bf*)pb;
      v8bf hi = *(const v8bf*)(pb + 16);
#pragma unroll
      for (int e = 0; e < 8; ++e) { bfrag[tI][e] = lo[e]; bfrag[tI][e + 8] = hi[e]; }
    }

#pragma unroll
    for (int mt = 0; mt < 2; ++mt)
#pragma unroll
      for (int nt = 0; nt < 4; ++nt)
        acc[mt][nt] = __builtin_amdgcn_wmma_f32_16x16x32_bf16(
            false, afrag[mt], false, bfrag[nt], (short)0, acc[mt][nt], false, false);

    __syncthreads();
  }

  // epilogue: C/D layout -> m = 8*kh + vgpr, n = lane&15
#pragma unroll
  for (int mt = 0; mt < 2; ++mt) {
#pragma unroll
    for (int nt = 0; nt < 4; ++nt) {
      int n = bn + wc * 64 + nt * 16 + lm;
      float bv = bias ? bias[n] : 0.f;
#pragma unroll
      for (int rr = 0; rr < 8; ++rr) {
        int m = bm + wr * 32 + mt * 16 + kh * 8 + rr;
        if (m < M) {
          float v = acc[mt][nt][rr] + bv;
          if (relu) v = fmaxf(v, 0.f);
          C[(long long)m * ldC + n] = v;
          if (Cbf) Cbf[(long long)m * N + n] = (__bf16)v;
        }
      }
    }
  }
}

// ---------------------------------------------------------------- launch
extern "C" void kernel_launch(void* const* d_in, const int* in_sizes, int n_in,
                              void* d_out, int out_size, void* d_ws, size_t ws_size,
                              hipStream_t stream) {
  (void)in_sizes; (void)n_in; (void)out_size; (void)ws_size;
  const int B = 2, S = 256, D = 1024, K = 8, V = 32000, F = 2048;
  const int I = S - K, M = B * I;

  const int*   x       = (const int*)  d_in[0];
  const float* embed_t = (const float*)d_in[1];
  const float* unembed = (const float*)d_in[2];
  const float* proj_w  = (const float*)d_in[3];
  const float* proj_b  = (const float*)d_in[4];
  const float* sa_wv   = (const float*)d_in[5];
  const float* sa_bv   = (const float*)d_in[6];
  const float* sa_wo   = (const float*)d_in[7];
  const float* sa_bo   = (const float*)d_in[8];
  const float* ln1_g   = (const float*)d_in[9];
  const float* ln1_b   = (const float*)d_in[10];
  const float* ca_wv   = (const float*)d_in[11];
  const float* ca_bv   = (const float*)d_in[12];
  const float* ca_wo   = (const float*)d_in[13];
  const float* ca_bo   = (const float*)d_in[14];
  const float* ln2_g   = (const float*)d_in[15];
  const float* ln2_b   = (const float*)d_in[16];
  const float* ff_w1   = (const float*)d_in[17];
  const float* ff_b1   = (const float*)d_in[18];
  const float* ff_w2   = (const float*)d_in[19];
  const float* ff_b2   = (const float*)d_in[20];
  const float* ln3_g   = (const float*)d_in[21];
  const float* ln3_b   = (const float*)d_in[22];
  float* out = (float*)d_out;

  char* ws = (char*)d_ws;
  size_t off = 0;
  auto alloc = [&](size_t bytes) -> void* {
    void* p = ws + off;
    off += (bytes + 255) & ~(size_t)255;
    return p;
  };

  float*  emb    = (float*) alloc((size_t)B * S * D * 4);
  __bf16* merged = (__bf16*)alloc((size_t)M * 2 * D * 2);
  float*  xp     = (float*) alloc((size_t)M * D * 4);
  __bf16* xpbf   = (__bf16*)alloc((size_t)M * D * 2);
  float*  t1     = (float*) alloc((size_t)M * D * 4);
  __bf16* t1bf   = (__bf16*)alloc((size_t)M * D * 2);
  float*  tacc   = (float*) alloc((size_t)M * D * 4);
  float*  x1     = (float*) alloc((size_t)M * D * 4);
  __bf16* x1bf   = (__bf16*)alloc((size_t)M * D * 2);
  float*  x2     = (float*) alloc((size_t)M * D * 4);
  __bf16* x2bf   = (__bf16*)alloc((size_t)M * D * 2);
  float*  t2     = (float*) alloc((size_t)M * F * 4);
  __bf16* t2bf   = (__bf16*)alloc((size_t)M * F * 2);
  float*  x3     = (float*) alloc((size_t)M * D * 4);
  __bf16* x3bf   = (__bf16*)alloc((size_t)M * D * 2);

  mtp_gather<<<B * S, 256, 0, stream>>>(x, embed_t, emb, D);

  dim3 blk(256);
  int mb = (M + 127) / 128;   // 4

  for (int k = 0; k < K; ++k) {
    const float* hsrc = (k == 0) ? emb : x3;
    int hskip = (k == 0) ? (S - I) : 0;
    mtp_rms_merge<<<M, 256, 0, stream>>>(hsrc, hskip, emb, S - I, k + 1, merged, I, D);

    // xp = merged @ proj_w[k].T + proj_b[k]
    mtp_gemm<<<dim3(D / 128, mb), blk, 0, stream>>>(merged, proj_w + (size_t)k * D * 2 * D,
        proj_b + (size_t)k * D, xp, xpbf, M, D, 2 * D, D, 0);
    // sa = (xp @ wv.T + bv) @ wo.T + bo
    mtp_gemm<<<dim3(D / 128, mb), blk, 0, stream>>>(xpbf, sa_wv + (size_t)k * D * D,
        sa_bv + (size_t)k * D, t1, t1bf, M, D, D, D, 0);
    mtp_gemm<<<dim3(D / 128, mb), blk, 0, stream>>>(t1bf, sa_wo + (size_t)k * D * D,
        sa_bo + (size_t)k * D, tacc, (__bf16*)nullptr, M, D, D, D, 0);
    mtp_ln_res<<<M, 256, 0, stream>>>(xp, tacc, ln1_g + (size_t)k * D, ln1_b + (size_t)k * D,
        x1, x1bf, D);
    // ca
    mtp_gemm<<<dim3(D / 128, mb), blk, 0, stream>>>(xpbf, ca_wv + (size_t)k * D * D,
        ca_bv + (size_t)k * D, t1, t1bf, M, D, D, D, 0);
    mtp_gemm<<<dim3(D / 128, mb), blk, 0, stream>>>(t1bf, ca_wo + (size_t)k * D * D,
        ca_bo + (size_t)k * D, tacc, (__bf16*)nullptr, M, D, D, D, 0);
    mtp_ln_res<<<M, 256, 0, stream>>>(x1, tacc, ln2_g + (size_t)k * D, ln2_b + (size_t)k * D,
        x2, x2bf, D);
    // ff
    mtp_gemm<<<dim3(F / 128, mb), blk, 0, stream>>>(x2bf, ff_w1 + (size_t)k * F * D,
        ff_b1 + (size_t)k * F, t2, t2bf, M, F, D, F, 1);
    mtp_gemm<<<dim3(D / 128, mb), blk, 0, stream>>>(t2bf, ff_w2 + (size_t)k * D * F,
        ff_b2 + (size_t)k * D, tacc, (__bf16*)nullptr, M, D, F, D, 0);
    mtp_ln_res<<<M, 256, 0, stream>>>(x2, tacc, ln3_g + (size_t)k * D, ln3_b + (size_t)k * D,
        x3, x3bf, D);
    // logits[:, :, k, :] = x3 @ unembed.T  (write straight into (B,I,K,V) layout)
    mtp_gemm<<<dim3(V / 128, mb), blk, 0, stream>>>(x3bf, unembed, (const float*)nullptr,
        out + (size_t)k * V, (__bf16*)nullptr, M, V, D, (long long)K * V, 0);
  }
}
